// MultiHeadAttention_88252987998666
// MI455X (gfx1250) — matrix operand order repaired and checked by reading, not compiled, not run on hardware
//
#include <hip/hip_runtime.h>
#include <hip/hip_bf16.h>

typedef __attribute__((ext_vector_type(16))) _Float16 v16h;
typedef __attribute__((ext_vector_type(8)))  float    v8f;
typedef __attribute__((ext_vector_type(2)))  _Float16 h2;
typedef __attribute__((ext_vector_type(4)))  _Float16 h4;
typedef __attribute__((ext_vector_type(8)))  _Float16 h8;
typedef __attribute__((ext_vector_type(4)))  unsigned int u32x4;
typedef __attribute__((ext_vector_type(8)))  int      i32x8;
typedef __attribute__((ext_vector_type(4)))  int      i32x4;

#define WMMA_F16(A, B, C) \
    __builtin_amdgcn_wmma_f32_16x16x32_f16(false, (A), false, (B), (short)0, (C), false, false)

#if defined(__has_builtin)
#  if __has_builtin(__builtin_amdgcn_tensor_load_to_lds) && \
      __has_builtin(__builtin_amdgcn_s_wait_tensorcnt)
#    define HAVE_TDM 1
#  endif
#endif
#ifndef HAVE_TDM
#  define HAVE_TDM 0
#endif

__device__ __forceinline__ float rmax16(float v) {
    v = fmaxf(v, __shfl_xor(v, 8, 32));
    v = fmaxf(v, __shfl_xor(v, 4, 32));
    v = fmaxf(v, __shfl_xor(v, 2, 32));
    v = fmaxf(v, __shfl_xor(v, 1, 32));
    return v;
}
__device__ __forceinline__ float rsum16(float v) {
    v += __shfl_xor(v, 8, 32);
    v += __shfl_xor(v, 4, 32);
    v += __shfl_xor(v, 2, 32);
    v += __shfl_xor(v, 1, 32);
    return v;
}

#if HAVE_TDM
// ---------------------------------------------------------------------------
// TDM 2D tile load: rows x elems0 f16 tile, global row stride = stride_elems,
// into LDS at lds_off with 16B padding after each 128B row (stride 144B = 72
// f16, matching SP below).  D# packing per CDNA5 ISA.
// ---------------------------------------------------------------------------
__device__ __forceinline__ void tdm_load_2d_f16(unsigned int lds_off,
                                                const void* gptr,
                                                unsigned int elems0,
                                                unsigned int rows,
                                                unsigned long long stride_elems)
{
    unsigned long long ga = (unsigned long long)(uintptr_t)gptr;
    u32x4 g0;
    g0[0] = 1u;                                   // count=1 valid descriptor
    g0[1] = lds_off;                              // lds_addr (bytes)
    g0[2] = (unsigned int)ga;                     // global_addr[31:0]
    g0[3] = (unsigned int)((ga >> 32) & 0x1FFFFFFu) | (2u << 30);  // [56:32]|type=2
    i32x8 g1;
    // data_size=2B (code 1), pad_enable, pad_interval=32 DW (code 4),
    // pad_amount=4 DW (code 3); workgroup_mask=0 (not in a cluster).
    g1[0] = (int)((1u << 16) | (1u << 20) | (4u << 22) | (3u << 25));
    g1[1] = (int)(elems0 << 16);                  // tensor_dim0[15:0] -> bits[63:48]
    g1[2] = (int)((elems0 >> 16) | (rows << 16)); // tensor_dim0[31:16] | tensor_dim1[15:0]
    g1[3] = (int)((rows >> 16) | (elems0 << 16)); // tensor_dim1[31:16] | tile_dim0
    g1[4] = (int)rows;                            // tile_dim1 | tile_dim2=0
    g1[5] = (int)(unsigned int)stride_elems;      // tensor_dim0_stride[31:0]
    g1[6] = (int)(unsigned int)(stride_elems >> 32);  // stride[47:32] | dim1_stride lo
    g1[7] = 0;
    i32x4 z4 = {0, 0, 0, 0};
#if __clang_major__ >= 23
    i32x8 z8 = {0, 0, 0, 0, 0, 0, 0, 0};
    __builtin_amdgcn_tensor_load_to_lds(g0, g1, z4, z4, z8, 0);
#else
    __builtin_amdgcn_tensor_load_to_lds(g0, g1, z4, z4, 0);
#endif
}
#endif  // HAVE_TDM

// ---------------------------------------------------------------------------
// GEMM: C[M,N] = A[M,K] @ W[N,K]^T + bias[N]
// A, W, bias f32; OutT = f16 (QKV) or f32 (final).  If VT, the f16 output is
// written per-head transposed: C[b*N*Sdim + n*Sdim + s]  (contiguous in s).
// Block 256 threads (8 waves), tile 128x64x32, wave = 32x32 via 2x2 WMMA.
// ---------------------------------------------------------------------------
constexpr int BM = 128, BN = 64, BK = 32;
constexpr int LDA = BK + 8;   // padded f16 stride (40)

template <typename OutT, bool VT>
__global__ __launch_bounds__(256) void gemm_xwT_bias(
    const float* __restrict__ A, const float* __restrict__ W,
    const float* __restrict__ bias, OutT* __restrict__ C,
    int M, int N, int K, int Sdim)
{
    __shared__ _Float16 As[BM * LDA];
    __shared__ _Float16 Bs[BN * LDA];

    const int tid  = threadIdx.x;
    const int lane = tid & 31;
    const int wave = tid >> 5;
    const int ln16 = lane & 15;
    const int hi   = (lane >> 4) & 1;

    const int bm = blockIdx.y * BM;
    const int bn = blockIdx.x * BN;
    const int wm = (wave & 3) * 32;
    const int wn = (wave >> 2) * 32;

    v8f acc[2][2];
#pragma unroll
    for (int mi = 0; mi < 2; ++mi)
#pragma unroll
        for (int ni = 0; ni < 2; ++ni)
#pragma unroll
            for (int r = 0; r < 8; ++r) acc[mi][ni][r] = 0.0f;

    for (int k0 = 0; k0 < K; k0 += BK) {
        // Stage A tile (128x32 f32 -> f16) with b128 loads.
#pragma unroll
        for (int i = 0; i < 4; ++i) {
            int idx = tid + i * 256;
            int r = idx >> 3, c4 = idx & 7;
            float4 v = *(const float4*)(A + (size_t)(bm + r) * K + k0 + c4 * 4);
            h4 hv; hv.x = (_Float16)v.x; hv.y = (_Float16)v.y;
            hv.z = (_Float16)v.z; hv.w = (_Float16)v.w;
            *(h4*)(&As[r * LDA + c4 * 4]) = hv;
        }
#pragma unroll
        for (int i = 0; i < 2; ++i) {
            int idx = tid + i * 256;
            int r = idx >> 3, c4 = idx & 7;
            float4 v = *(const float4*)(W + (size_t)(bn + r) * K + k0 + c4 * 4);
            h4 hv; hv.x = (_Float16)v.x; hv.y = (_Float16)v.y;
            hv.z = (_Float16)v.z; hv.w = (_Float16)v.w;
            *(h4*)(&Bs[r * LDA + c4 * 4]) = hv;
        }
        // Prefetch next K-slab (global_prefetch_b8).
        if (k0 + BK < K) {
            if (tid < BM)
                __builtin_prefetch(A + (size_t)(bm + tid) * K + k0 + BK, 0, 1);
            else if (tid < BM + BN)
                __builtin_prefetch(W + (size_t)(bn + tid - BM) * K + k0 + BK, 0, 1);
        }
        __syncthreads();

        v16h af[2], bf[2];
#pragma unroll
        for (int mi = 0; mi < 2; ++mi) {
            const _Float16* p = &As[(wm + mi * 16 + ln16) * LDA + hi * 8];
#pragma unroll
            for (int e = 0; e < 16; e += 2) {
                int kk = (e < 8) ? e : (e + 8);
                h2 t = *(const h2*)(p + kk);
                af[mi][e] = t.x; af[mi][e + 1] = t.y;
            }
        }
#pragma unroll
        for (int ni = 0; ni < 2; ++ni) {
            const _Float16* p = &Bs[(wn + ni * 16 + ln16) * LDA + hi * 8];
#pragma unroll
            for (int e = 0; e < 16; e += 2) {
                int kk = (e < 8) ? e : (e + 8);
                h2 t = *(const h2*)(p + kk);
                bf[ni][e] = t.x; bf[ni][e + 1] = t.y;
            }
        }
#pragma unroll
        for (int mi = 0; mi < 2; ++mi)
#pragma unroll
            for (int ni = 0; ni < 2; ++ni)
                acc[mi][ni] = WMMA_F16(af[mi], bf[ni], acc[mi][ni]);
        __syncthreads();
    }

#pragma unroll
    for (int mi = 0; mi < 2; ++mi)
#pragma unroll
        for (int ni = 0; ni < 2; ++ni) {
            int n = bn + wn + ni * 16 + ln16;
            float bv = bias[n];
#pragma unroll
            for (int r = 0; r < 8; ++r) {
                int m = bm + wm + mi * 16 + r + hi * 8;
                float val = acc[mi][ni][r] + bv;
                if (VT) {
                    int bq = m / Sdim, sq = m % Sdim;
                    C[(size_t)bq * N * Sdim + (size_t)n * Sdim + sq] = (OutT)val;
                } else {
                    C[(size_t)m * N + n] = (OutT)val;
                }
            }
        }
}

// ---------------------------------------------------------------------------
// Flash attention over one (b, h), 128 q-rows per block (8 waves x 16 rows).
// Qh/Kh: f16 [B*S, E]; Vt: f16 [B, E, S] per-head transposed; O: f32 [B*S, E].
// K and V tiles are DMA'd into LDS by the Tensor Data Mover (wave 0), with
// TDM padding producing the 72-element LDS stride.
// ---------------------------------------------------------------------------
constexpr int KT = 64;   // keys per tile
constexpr int DH = 64;   // head dim
constexpr int SP = 72;   // padded LDS stride (TDM: 32 DW data + 4 DW pad)

__global__ __launch_bounds__(256) void attn_flash(
    const _Float16* __restrict__ Qh, const _Float16* __restrict__ Kh,
    const _Float16* __restrict__ Vt, float* __restrict__ O,
    int Bn, int Sn, int Hn)
{
    __shared__ _Float16 Ks[KT * SP];        // [key][d]
    __shared__ _Float16 Vs[DH * SP];        // [d][key]
    __shared__ _Float16 Ps[8 * 16 * SP];    // per-wave P tile [16][64]

    const int E = Hn * DH;
    const int tid  = threadIdx.x;
    const int lane = tid & 31;
    const int wave = tid >> 5;
    const int ln16 = lane & 15;
    const int hi   = (lane >> 4) & 1;

    const int bh = blockIdx.y;
    const int b  = bh / Hn;
    const int h  = bh % Hn;
    const int q0 = blockIdx.x * 128 + wave * 16;
    const size_t base  = ((size_t)b * Sn) * E + (size_t)h * DH;        // K rows
    const size_t vbase = (size_t)b * E * Sn + (size_t)(h * DH) * Sn;   // Vt rows

    // Q fragments (A-matrix: M=16 q-rows, K=d), loaded once from global f16.
    v16h qf[2];
#pragma unroll
    for (int c = 0; c < 2; ++c) {
        const _Float16* qp = Qh + base + (size_t)(q0 + ln16) * E + c * 32 + hi * 8;
#pragma unroll
        for (int e = 0; e < 16; e += 2) {
            int kk = (e < 8) ? e : (e + 8);
            h2 t = *(const h2*)(qp + kk);
            qf[c][e] = t.x; qf[c][e + 1] = t.y;
        }
    }

    float mrow[8], lrow[8];
    v8f oacc[4];
#pragma unroll
    for (int r = 0; r < 8; ++r) { mrow[r] = -1.0e30f; lrow[r] = 0.0f; }
#pragma unroll
    for (int t = 0; t < 4; ++t)
#pragma unroll
        for (int r = 0; r < 8; ++r) oacc[t][r] = 0.0f;

    const float scale = 0.125f;   // 1/sqrt(64)

#if HAVE_TDM
    const unsigned int ks_off = (unsigned int)(uintptr_t)(&Ks[0]);
    const unsigned int vs_off = (unsigned int)(uintptr_t)(&Vs[0]);
#endif

    for (int kb = 0; kb < Sn; kb += KT) {
#if HAVE_TDM
        if (wave == 0) {
            // K tile: 64 keys x 64 d, row stride E elems.
            tdm_load_2d_f16(ks_off, Kh + base + (size_t)kb * E, DH, KT, (unsigned long long)E);
            // V tile (pre-transposed): 64 d x 64 keys, row stride S elems.
            tdm_load_2d_f16(vs_off, Vt + vbase + kb, KT, DH, (unsigned long long)Sn);
            __builtin_amdgcn_s_wait_tensorcnt(0);
        }
#else
#pragma unroll
        for (int i = 0; i < 2; ++i) {
            int idx = tid + i * 256;
            int r = idx >> 3, c8 = (idx & 7) * 8;
            h8 kv = *(const h8*)(Kh + base + (size_t)(kb + r) * E + c8);
            *(h8*)(&Ks[r * SP + c8]) = kv;
            h8 vv = *(const h8*)(Vt + vbase + (size_t)r * Sn + kb + c8);
            *(h8*)(&Vs[r * SP + c8]) = vv;
        }
#endif
        __syncthreads();

        // Scores: S = Q @ K^T  (4 key-tiles of 16, K-dim = 64 => 2 wmma each)
        v8f s[4];
        v16h kf;
#pragma unroll
        for (int t = 0; t < 4; ++t) {
#pragma unroll
            for (int r = 0; r < 8; ++r) s[t][r] = 0.0f;
#pragma unroll
            for (int c = 0; c < 2; ++c) {
                const _Float16* p = &Ks[(t * 16 + ln16) * SP + c * 32 + hi * 8];
#pragma unroll
                for (int e = 0; e < 16; e += 2) {
                    h2 v = *(const h2*)(p + ((e < 8) ? e : (e + 8)));
                    kf[e] = v.x; kf[e + 1] = v.y;
                }
                s[t] = WMMA_F16(qf[c], kf, s[t]);
            }
#pragma unroll
            for (int r = 0; r < 8; ++r) s[t][r] *= scale;
        }

        // Online softmax per row m = r + 8*hi; cols striped over 16 lanes.
#pragma unroll
        for (int r = 0; r < 8; ++r) {
            float mx = s[0][r];
#pragma unroll
            for (int t = 1; t < 4; ++t) mx = fmaxf(mx, s[t][r]);
            mx = rmax16(mx);
            float newm = fmaxf(mrow[r], mx);
            float corr = __expf(mrow[r] - newm);
            mrow[r] = newm;
            float ps = 0.0f;
            _Float16* pp = &Ps[(wave * 16 + r + hi * 8) * SP + ln16];
#pragma unroll
            for (int t = 0; t < 4; ++t) {
                float p = __expf(s[t][r] - newm);
                ps += p;
                pp[t * 16] = (_Float16)p;
            }
            ps = rsum16(ps);
            lrow[r] = lrow[r] * corr + ps;
#pragma unroll
            for (int t = 0; t < 4; ++t) oacc[t][r] *= corr;
        }

        // P fragments (same-wave LDS RAW kept in order by DS pipeline).
        v16h pf[2];
#pragma unroll
        for (int c = 0; c < 2; ++c) {
            const _Float16* p = &Ps[(wave * 16 + ln16) * SP + c * 32 + hi * 8];
#pragma unroll
            for (int e = 0; e < 16; e += 2) {
                int kk = (e < 8) ? e : (e + 8);
                h2 t = *(const h2*)(p + kk);
                pf[c][e] = t.x; pf[c][e + 1] = t.y;
            }
        }
        // O += P @ V   (4 d-tiles of 16, K-dim = 64 keys => 2 wmma each)
        v16h vf;
#pragma unroll
        for (int t = 0; t < 4; ++t)
#pragma unroll
            for (int c = 0; c < 2; ++c) {
                const _Float16* p = &Vs[(t * 16 + ln16) * SP + c * 32 + hi * 8];
#pragma unroll
                for (int e = 0; e < 16; e += 2) {
                    h2 v = *(const h2*)(p + ((e < 8) ? e : (e + 8)));
                    vf[e] = v.x; vf[e + 1] = v.y;
                }
                oacc[t] = WMMA_F16(pf[c], vf, oacc[t]);
            }
        __syncthreads();
    }

    // Finalize: divide by running row sums, store f32 [B*S, E].
#pragma unroll
    for (int r = 0; r < 8; ++r) {
        float inv = 1.0f / lrow[r];
        int row = q0 + r + hi * 8;
#pragma unroll
        for (int t = 0; t < 4; ++t) {
            int col = h * DH + t * 16 + ln16;
            O[((size_t)b * Sn + row) * E + col] = oacc[t][r] * inv;
        }
    }
}

// ---------------------------------------------------------------------------
// Launch: QKV projections (f16; V transposed per head) -> flash attention
// (TDM-staged tiles) -> output projection (f32).
// ---------------------------------------------------------------------------
extern "C" void kernel_launch(void* const* d_in, const int* in_sizes, int n_in,
                              void* d_out, int out_size, void* d_ws, size_t ws_size,
                              hipStream_t stream)
{
    (void)in_sizes; (void)n_in; (void)out_size; (void)ws_size;
    constexpr int B = 2, S = 2048, E = 1024, H = 16;
    constexpr size_t M = (size_t)B * S;   // 4096

    const float* q_x = (const float*)d_in[0];
    const float* k_x = (const float*)d_in[1];
    const float* v_x = (const float*)d_in[2];
    const float* q_w = (const float*)d_in[3];
    const float* q_b = (const float*)d_in[4];
    const float* k_w = (const float*)d_in[5];
    const float* k_b = (const float*)d_in[6];
    const float* v_w = (const float*)d_in[7];
    const float* v_b = (const float*)d_in[8];
    const float* o_w = (const float*)d_in[9];
    const float* o_b = (const float*)d_in[10];

    // Workspace: Qh/Kh/Vt (f16, 8MB each) + attn (f32, 16MB).
    _Float16* Qh = (_Float16*)d_ws;
    _Float16* Kh = Qh + M * E;
    _Float16* Vt = Kh + M * E;
    float*    At = (float*)(Vt + M * E);

    dim3 blk(256);
    dim3 gg(E / BN, (int)(M / BM));            // (16, 32)

    gemm_xwT_bias<_Float16, false><<<gg, blk, 0, stream>>>(q_x, q_w, q_b, Qh, (int)M, E, E, S);
    gemm_xwT_bias<_Float16, false><<<gg, blk, 0, stream>>>(k_x, k_w, k_b, Kh, (int)M, E, E, S);
    gemm_xwT_bias<_Float16, true ><<<gg, blk, 0, stream>>>(v_x, v_w, v_b, Vt, (int)M, E, E, S);

    dim3 ga(S / 128, B * H);                   // (16, 32)
    attn_flash<<<ga, blk, 0, stream>>>(Qh, Kh, Vt, At, B, S, H);

    gemm_xwT_bias<float, false><<<gg, blk, 0, stream>>>(At, o_w, o_b, (float*)d_out, (int)M, E, E, S);
}
